// AttentionLayer_40243843564115
// MI455X (gfx1250) — matrix operand order repaired and checked by reading, not compiled, not run on hardware
//
#include <hip/hip_runtime.h>
#include <hip/hip_bf16.h>

typedef __attribute__((ext_vector_type(16))) __bf16 v16bf;
typedef __attribute__((ext_vector_type(8)))  float  v8f;
typedef __attribute__((ext_vector_type(4)))  int    v4i;

#define Bn 4
#define Tn 2048
#define Cn 1024
#define Hn 16
#define Dn 64

#if defined(__gfx1250__) && __has_builtin(__builtin_amdgcn_global_load_async_to_lds_b128) && __has_builtin(__builtin_amdgcn_s_wait_asynccnt)
#define ASYNC_LDS 1
#else
#define ASYNC_LDS 0
#endif

__device__ __forceinline__ unsigned short f2bf(float f) {
    __bf16 h = (__bf16)f;                     // native v_cvt_*bf16* (RNE)
    return __builtin_bit_cast(unsigned short, h);
}

__device__ __forceinline__ void async_cp16(const unsigned short* g, unsigned short* l) {
#if ASYNC_LDS
    typedef __attribute__((address_space(1))) v4i GV4;
    typedef __attribute__((address_space(3))) v4i LV4;
    __builtin_amdgcn_global_load_async_to_lds_b128((GV4*)g, (LV4*)l, 0, 0);
#else
    *(uint4*)l = *(const uint4*)g;
#endif
}

__device__ __forceinline__ void async_fence() {
#if ASYNC_LDS
    __builtin_amdgcn_s_wait_asynccnt(0);
#endif
}

union FragB { v16bf v; uint4 q[2]; unsigned short s[16]; };
union Pack8 { uint4 v; unsigned short s[8]; };

// ---------------------------------------------------------------------------
// GEMM1: X[8192,1024] f32  @  Wqkv[1024,3072] f32  + bqkv  ->  q/k/v bf16
// planes laid out [B,H,T,D]. 128 threads (4 waves), 64x64 tile per block.
// ---------------------------------------------------------------------------
__global__ __launch_bounds__(128)
void qkv_gemm(const float* __restrict__ X, const float* __restrict__ W,
              const float* __restrict__ bias, unsigned short* __restrict__ qkv) {
    __shared__ __align__(16) unsigned short Abuf[2048];
    __shared__ __align__(16) unsigned short Bbuf[2048];
    const int tid = threadIdx.x, lane = tid & 31, w = tid >> 5;
    const int n0 = blockIdx.x * 64, m0 = blockIdx.y * 64;
    const int Nld = 3 * Cn;
    v8f acc[4] = {};

    for (int k0 = 0; k0 < Cn; k0 += 32) {
        __syncthreads();
        // stage A (64x32 f32 -> bf16 fragment order): 256 runs of 8
#pragma unroll
        for (int rr = 0; rr < 2; ++rr) {
            int rid = tid + rr * 128;
            int r = rid >> 2, kq = rid & 3;
            const float4* src = (const float4*)&X[(m0 + r) * Cn + k0 + kq * 8];
            float4 f0 = src[0], f1 = src[1];
            Pack8 p;
            p.s[0]=f2bf(f0.x); p.s[1]=f2bf(f0.y); p.s[2]=f2bf(f0.z); p.s[3]=f2bf(f0.w);
            p.s[4]=f2bf(f1.x); p.s[5]=f2bf(f1.y); p.s[6]=f2bf(f1.z); p.s[7]=f2bf(f1.w);
            int off = (r >> 4) * 512 + ((r & 15) + ((kq & 1) << 4)) * 16 + ((kq >> 1) << 3);
            *(uint4*)&Abuf[off] = p.v;
        }
        // stage B (32x64): coalesced f32 reads, scattered bf16 LDS writes
#pragma unroll
        for (int rr = 0; rr < 2; ++rr) {
            int rid = tid + rr * 128;
            int k = rid >> 3, n8 = (rid & 7) * 8;
            const float4* src = (const float4*)&W[(k0 + k) * Nld + n0 + n8];
            float4 f0 = src[0], f1 = src[1];
            float vv[8] = {f0.x, f0.y, f0.z, f0.w, f1.x, f1.y, f1.z, f1.w};
#pragma unroll
            for (int i = 0; i < 8; ++i) {
                int n = n8 + i;
                Bbuf[(n >> 4) * 512 + ((n & 15) + (((k >> 3) & 1) << 4)) * 16 + ((k >> 4) << 3) + (k & 7)] = f2bf(vv[i]);
            }
        }
        __syncthreads();
        FragB a, bb[4];
        a.q[0] = *(const uint4*)&Abuf[w * 512 + lane * 16];
        a.q[1] = *(const uint4*)&Abuf[w * 512 + lane * 16 + 8];
#pragma unroll
        for (int jn = 0; jn < 4; ++jn) {
            bb[jn].q[0] = *(const uint4*)&Bbuf[jn * 512 + lane * 16];
            bb[jn].q[1] = *(const uint4*)&Bbuf[jn * 512 + lane * 16 + 8];
        }
#pragma unroll
        for (int jn = 0; jn < 4; ++jn)
            acc[jn] = __builtin_amdgcn_wmma_f32_16x16x32_bf16(
                false, a.v, false, bb[jn].v, (short)0, acc[jn], false, false);
    }
    // epilogue: bias add, scatter to q/k/v [B,H,T,D] bf16
    const int nloc = lane & 15, half = lane >> 4;
    const int mrow0 = m0 + w * 16 + half * 8;
#pragma unroll
    for (int jn = 0; jn < 4; ++jn) {
        int col = n0 + jn * 16 + nloc;
        float bv = bias[col];
        int which = col >> 10;
        int c = col & (Cn - 1);
        int hh = c >> 6, dd = c & 63;
        unsigned short* dst = qkv + (size_t)which * ((size_t)Bn * Hn * Tn * Dn);
#pragma unroll
        for (int j = 0; j < 8; ++j) {
            int row = mrow0 + j;
            int bb2 = row >> 11, tt = row & (Tn - 1);
            dst[(size_t)(((bb2 * Hn) + hh) * Tn + tt) * Dn + dd] = f2bf(acc[jn][j] + bv);
        }
    }
}

// ---------------------------------------------------------------------------
// Flash attention: per block = one (b,h) and 64 q-rows; 4 waves x 16 rows.
// kv chunks of 32, online softmax, bf16 WMMA, causal mask, scale 1/32.
// K chunk staged via async LDS copy (fragment-order permute of 16B blocks).
// ---------------------------------------------------------------------------
__global__ __launch_bounds__(128)
void flash_attn(const unsigned short* __restrict__ Qb,
                const unsigned short* __restrict__ Kb,
                const unsigned short* __restrict__ Vb,
                unsigned short* __restrict__ X2) {
    __shared__ __align__(16) unsigned short Kbuf[2048];   // 4 B-frags of K^T chunk
    __shared__ __align__(16) unsigned short Vbuf[2048];   // 4 B-frags of V chunk
    __shared__ __align__(16) unsigned short Pbuf[2048];   // per-wave 16x32 bounce
    const int tid = threadIdx.x, lane = tid & 31, w = tid >> 5;
    const int q0 = blockIdx.x * 64;
    const int bh = blockIdx.y;
    const int b = bh >> 4, h = bh & 15;
    const int nloc = lane & 15, half = lane >> 4;
    const int qw = q0 + w * 16;
    const size_t headbase = (size_t)bh * Tn * Dn;
    const float NEG_INF = -__builtin_inff();

    // Q fragments (A layout) straight from global
    FragB qf[2];
    {
        const unsigned short* qp = Qb + headbase + (size_t)(qw + nloc) * Dn;
#pragma unroll
        for (int jd = 0; jd < 2; ++jd) {
            const unsigned short* p = qp + jd * 32 + half * 8;
            qf[jd].q[0] = *(const uint4*)p;
            qf[jd].q[1] = *(const uint4*)(p + 16);
        }
    }

    v8f o[4] = {};
    float mr[8], lr[8];
#pragma unroll
    for (int j = 0; j < 8; ++j) { mr[j] = NEG_INF; lr[j] = 0.f; }

    const int kvend = q0 + 64;
    for (int kvc = 0; kvc < kvend; kvc += 32) {
        __syncthreads();
        // --- K chunk: 256 16B-block permuted copies (async -> LDS) ---------
#pragma unroll
        for (int rr = 0; rr < 2; ++rr) {
            int rid = tid + rr * 128;
            int tl = rid >> 3, d8 = (rid & 7) * 8;
            const unsigned short* g = Kb + headbase + (size_t)(kvc + tl) * Dn + d8;
            int frag = (tl >> 4) * 2 + (d8 >> 5);
            int loff = frag * 512 + ((tl & 15) + (((d8 >> 3) & 1) << 4)) * 16 + (((d8 >> 4) & 1) << 3);
            async_cp16(g, &Kbuf[loff]);
        }
        // --- V chunk: pair two kv rows -> one b32 LDS store ----------------
        {
            int t0 = (tid >> 3) * 2, d0 = (tid & 7) * 8;
            const unsigned short* src = Vb + headbase + (size_t)(kvc + t0) * Dn + d0;
            Pack8 a, c;
            a.v = *(const uint4*)src;
            c.v = *(const uint4*)(src + Dn);
            int basef = (d0 >> 4) * 512 + (((t0 >> 3) & 1) << 4) * 16 + ((t0 >> 4) << 3) + (t0 & 7);
#pragma unroll
            for (int i = 0; i < 8; ++i) {
                unsigned int pair = (unsigned int)a.s[i] | ((unsigned int)c.s[i] << 16);
                *(unsigned int*)&Vbuf[basef + ((d0 + i) & 15) * 16] = pair;
            }
        }
        async_fence();
        __syncthreads();

        // S = Q K^T : preload all 4 K fragments, then chain WMMAs
        FragB kf[4];
#pragma unroll
        for (int f = 0; f < 4; ++f) {
            kf[f].q[0] = *(const uint4*)&Kbuf[f * 512 + lane * 16];
            kf[f].q[1] = *(const uint4*)&Kbuf[f * 512 + lane * 16 + 8];
        }
        v8f s[2] = {};
        s[0] = __builtin_amdgcn_wmma_f32_16x16x32_bf16(false, qf[0].v, false, kf[0].v, (short)0, s[0], false, false);
        s[0] = __builtin_amdgcn_wmma_f32_16x16x32_bf16(false, qf[1].v, false, kf[1].v, (short)0, s[0], false, false);
        s[1] = __builtin_amdgcn_wmma_f32_16x16x32_bf16(false, qf[0].v, false, kf[2].v, (short)0, s[1], false, false);
        s[1] = __builtin_amdgcn_wmma_f32_16x16x32_bf16(false, qf[1].v, false, kf[3].v, (short)0, s[1], false, false);

        // scale (1/sqrt(C)=1/32) + causal mask
#pragma unroll
        for (int jn = 0; jn < 2; ++jn) {
            int kvn = kvc + jn * 16 + nloc;
#pragma unroll
            for (int j = 0; j < 8; ++j) {
                int qrow = qw + half * 8 + j;
                float val = s[jn][j] * 0.03125f;
                s[jn][j] = (kvn > qrow) ? NEG_INF : val;
            }
        }

        // online softmax; write exp(P) bf16 into per-wave LDS row-major 16x32
        unsigned short* Pw = Pbuf + w * 512;
#pragma unroll
        for (int j = 0; j < 8; ++j) {
            float mx = fmaxf(s[0][j], s[1][j]);
#pragma unroll
            for (int off = 1; off < 16; off <<= 1)
                mx = fmaxf(mx, __shfl_xor(mx, off, 32));
            float mnew = fmaxf(mr[j], mx);
            float alpha = __expf(mr[j] - mnew);
            mr[j] = mnew;
            float p0 = __expf(s[0][j] - mnew);
            float p1 = __expf(s[1][j] - mnew);
            float rs = p0 + p1;
#pragma unroll
            for (int off = 1; off < 16; off <<= 1)
                rs += __shfl_xor(rs, off, 32);
            lr[j] = lr[j] * alpha + rs;
#pragma unroll
            for (int jd = 0; jd < 4; ++jd) o[jd][j] *= alpha;
            int m = half * 8 + j;
            Pw[m * 32 + nloc]      = f2bf(p0);
            Pw[m * 32 + 16 + nloc] = f2bf(p1);
        }

        // reload P as A-fragment; preload V frags; O += P V
        FragB pf, vf[4];
        pf.q[0] = *(const uint4*)&Pw[nloc * 32 + half * 8];
        pf.q[1] = *(const uint4*)&Pw[nloc * 32 + 16 + half * 8];
#pragma unroll
        for (int jd = 0; jd < 4; ++jd) {
            vf[jd].q[0] = *(const uint4*)&Vbuf[jd * 512 + lane * 16];
            vf[jd].q[1] = *(const uint4*)&Vbuf[jd * 512 + lane * 16 + 8];
        }
#pragma unroll
        for (int jd = 0; jd < 4; ++jd)
            o[jd] = __builtin_amdgcn_wmma_f32_16x16x32_bf16(
                false, pf.v, false, vf[jd].v, (short)0, o[jd], false, false);
    }

    // epilogue: normalize and store in scrambled (H,T,B,D) flat order, bf16
#pragma unroll
    for (int jd = 0; jd < 4; ++jd) {
        int d = jd * 16 + nloc;
#pragma unroll
        for (int j = 0; j < 8; ++j) {
            int qrow = qw + half * 8 + j;
            float val = o[jd][j] / lr[j];
            size_t flat = ((size_t)(h * Tn + qrow) * Bn + b) * Dn + d;
            X2[flat] = f2bf(val);
        }
    }
}

// ---------------------------------------------------------------------------
// GEMM2: X2[8192,1024] bf16 @ Wout[1024,1024] f32 + bout -> d_out f32
// A tile staged via async LDS copy (16B-block fragment permute).
// ---------------------------------------------------------------------------
__global__ __launch_bounds__(128)
void out_gemm(const unsigned short* __restrict__ X2, const float* __restrict__ W,
              const float* __restrict__ bias, float* __restrict__ out) {
    __shared__ __align__(16) unsigned short Abuf[2048];
    __shared__ __align__(16) unsigned short Bbuf[2048];
    const int tid = threadIdx.x, lane = tid & 31, w = tid >> 5;
    const int n0 = blockIdx.x * 64, m0 = blockIdx.y * 64;
    v8f acc[4] = {};

    for (int k0 = 0; k0 < Cn; k0 += 32) {
        __syncthreads();
        // stage A (bf16 already): 256 16B-block permuted copies (async)
#pragma unroll
        for (int rr = 0; rr < 2; ++rr) {
            int rid = tid + rr * 128;
            int r = rid >> 2, kq = rid & 3;
            const unsigned short* g = X2 + (size_t)(m0 + r) * Cn + k0 + kq * 8;
            int loff = (r >> 4) * 512 + ((r & 15) + ((kq & 1) << 4)) * 16 + ((kq >> 1) << 3);
            async_cp16(g, &Abuf[loff]);
        }
        // stage B (32x64 of Wout, f32 -> bf16 fragment order)
#pragma unroll
        for (int rr = 0; rr < 2; ++rr) {
            int rid = tid + rr * 128;
            int k = rid >> 3, n8 = (rid & 7) * 8;
            const float4* src = (const float4*)&W[(k0 + k) * Cn + n0 + n8];
            float4 f0 = src[0], f1 = src[1];
            float vv[8] = {f0.x, f0.y, f0.z, f0.w, f1.x, f1.y, f1.z, f1.w};
#pragma unroll
            for (int i = 0; i < 8; ++i) {
                int n = n8 + i;
                Bbuf[(n >> 4) * 512 + ((n & 15) + (((k >> 3) & 1) << 4)) * 16 + ((k >> 4) << 3) + (k & 7)] = f2bf(vv[i]);
            }
        }
        async_fence();
        __syncthreads();
        FragB a, bb[4];
        a.q[0] = *(const uint4*)&Abuf[w * 512 + lane * 16];
        a.q[1] = *(const uint4*)&Abuf[w * 512 + lane * 16 + 8];
#pragma unroll
        for (int jn = 0; jn < 4; ++jn) {
            bb[jn].q[0] = *(const uint4*)&Bbuf[jn * 512 + lane * 16];
            bb[jn].q[1] = *(const uint4*)&Bbuf[jn * 512 + lane * 16 + 8];
        }
#pragma unroll
        for (int jn = 0; jn < 4; ++jn)
            acc[jn] = __builtin_amdgcn_wmma_f32_16x16x32_bf16(
                false, a.v, false, bb[jn].v, (short)0, acc[jn], false, false);
    }
    const int nloc = lane & 15, half = lane >> 4;
    const int mrow0 = m0 + w * 16 + half * 8;
#pragma unroll
    for (int jn = 0; jn < 4; ++jn) {
        int col = n0 + jn * 16 + nloc;
        float bv = bias[col];
#pragma unroll
        for (int j = 0; j < 8; ++j)
            out[(size_t)(mrow0 + j) * Cn + col] = acc[jn][j] + bv;
    }
}

extern "C" void kernel_launch(void* const* d_in, const int* in_sizes, int n_in,
                              void* d_out, int out_size, void* d_ws, size_t ws_size,
                              hipStream_t stream) {
    (void)in_sizes; (void)n_in; (void)out_size; (void)ws_size;
    const float* X    = (const float*)d_in[0];
    // d_in[1] = causal mask (implicit in kernel, unused)
    const float* Wqkv = (const float*)d_in[2];
    const float* bqkv = (const float*)d_in[3];
    const float* Wout = (const float*)d_in[4];
    const float* bout = (const float*)d_in[5];
    float* out = (float*)d_out;

    unsigned short* ws = (unsigned short*)d_ws;
    const size_t PL = (size_t)Bn * Hn * Tn * Dn;   // 8,388,608 elems per plane
    unsigned short* qkv = ws;                      // q,k,v planes (bf16)
    unsigned short* Qb  = ws;
    unsigned short* Kb  = ws + PL;
    unsigned short* Vb  = ws + 2 * PL;
    unsigned short* X2  = ws + 3 * PL;             // scrambled attn out (bf16)

    dim3 blk(128);
    qkv_gemm <<<dim3(48, 128), blk, 0, stream>>>(X, Wqkv, bqkv, qkv);
    flash_attn<<<dim3(32, 64),  blk, 0, stream>>>(Qb, Kb, Vb, X2);
    out_gemm <<<dim3(16, 128), blk, 0, stream>>>(X2, Wout, bout, out);
}
